// Embedding_52123723104389
// MI455X (gfx1250) — matrix operand order repaired and checked by reading, not compiled, not run on hardware
//
#include <hip/hip_runtime.h>
#include <math.h>

// ---------------- CDNA5 (gfx1250) types ----------------
typedef __attribute__((ext_vector_type(16))) __bf16 v16bf;
typedef __attribute__((ext_vector_type(8)))  float  v8f;
typedef __attribute__((ext_vector_type(4)))  float  f4_t;
typedef __attribute__((ext_vector_type(4)))  int    v4i;

typedef __attribute__((address_space(3))) char lds_char_t;

#define N_ROWS       65536
#define WAVES        4                     // 4 waves/block (LDS budget)
#define ABUF_USHORTS 4096                  // 8 kchunks * 32 lanes * 16 bf16 (A-fragment layout)
#define ABUF_BYTES   8192
#define EBUF_BYTES   (128 * 48)            // e1 stash: 128 cols * 48B
#define WAVE_SCR     (ABUF_BYTES + EBUF_BYTES)       // 14336 B per wave
#define WGT_BYTES    262144                // all weights, bf16, tile-swizzled
#define SMEM_TOTAL   (WGT_BYTES + WAVES * WAVE_SCR)  // 319488 <= 327680 (320KB/WGP)

#define W1_OFS 0        // ushort offsets: 16nt*4kc*512
#define W2_OFS 32768    // 16nt*8kc*512
#define W3_OFS 98304    //  8nt*8kc*512

#if defined(__has_builtin)
# if __has_builtin(__builtin_amdgcn_global_load_async_to_lds_b128)
#  define ASYNC_BUILTIN 1
# endif
#endif

// ---------------- scalar helpers ----------------
__device__ __forceinline__ unsigned short bf16_rne_bits(float f) {  // prep kernel only
  unsigned int u = __float_as_uint(f);
  u += 0x7FFFu + ((u >> 16) & 1u);
  return (unsigned short)(u >> 16);
}
__device__ __forceinline__ float bf16_to_f32(unsigned int h) {
  return __uint_as_float(h << 16);
}
__device__ __forceinline__ unsigned pack_bf16x2(float a, float b) {  // native cvt + pack
  union { __bf16 h[2]; unsigned u; } t;
  t.h[0] = (__bf16)a; t.h[1] = (__bf16)b;
  return t.u;
}
// Branchless SELU: selu(x) = l*max(x,0) + l*a*(exp(min(x,0)) - 1).
// exp(0)-1 == 0 kills the negative arm for x>0; no EXEC-mask churn.
__device__ __forceinline__ float selu_f(float x) {
  const float l  = 1.0507009873554805f;
  const float la = 1.7580993408473766f;   // l * 1.6732632423543772
  return l * fmaxf(x, 0.f) + la * (__expf(fminf(x, 0.f)) - 1.f);
}

// A fragment from a global f32 row (layer-1 input), native f32->bf16 cvt.
// 16-bit A layout (ISA 7.12.2): lanes 0-15 hold K {0-7,16-23}, lanes 16-31 hold K {8-15,24-31}.
__device__ __forceinline__ v16bf a_frag_from_f32(const float* rowp, int kbase, int sel) {
  const f4_t* p0 = (const f4_t*)(rowp + kbase + sel);
  const f4_t* p1 = (const f4_t*)(rowp + kbase + 16 + sel);
  f4_t f0 = __builtin_nontemporal_load(p0);
  f4_t f1 = __builtin_nontemporal_load(p0 + 1);
  f4_t f2 = __builtin_nontemporal_load(p1);
  f4_t f3 = __builtin_nontemporal_load(p1 + 1);
  v16bf a;
  a[0] = (__bf16)f0.x; a[1] = (__bf16)f0.y; a[2]  = (__bf16)f0.z; a[3]  = (__bf16)f0.w;
  a[4] = (__bf16)f1.x; a[5] = (__bf16)f1.y; a[6]  = (__bf16)f1.z; a[7]  = (__bf16)f1.w;
  a[8] = (__bf16)f2.x; a[9] = (__bf16)f2.y; a[10] = (__bf16)f2.z; a[11] = (__bf16)f2.w;
  a[12]= (__bf16)f3.x; a[13]= (__bf16)f3.y; a[14] = (__bf16)f3.z; a[15] = (__bf16)f3.w;
  return a;
}

// one 16x16 output tile: D(f32) = sum_kc A[kc] x B[kc] + bias
// wblk_lane already includes this lane's 32B slot; fragments are single aligned v16bf loads.
template <int KC>
__device__ __forceinline__ v8f mlp_tile(const v16bf* A, const __bf16* wblk_lane, float bv) {
  v8f c;
#pragma unroll
  for (int i = 0; i < 8; i++) c[i] = bv;
#pragma unroll
  for (int kc = 0; kc < KC; kc++) {
    v16bf b = *(const v16bf*)(wblk_lane + kc * 512);
    c = __builtin_amdgcn_wmma_f32_16x16x32_bf16(false, A[kc], false, b,
                                                (short)0, c, false, false);
  }
  return c;
}

// full 3-layer embed for one 16-row tile, one wave. SECOND accumulates dot/norms vs e1 stash.
template <bool SECOND>
__device__ __forceinline__ void embed_pass(const float* __restrict__ S, int row0,
                                           const __bf16* wlane,      // wgt + lane*16
                                           __bf16* abuf, unsigned int* ebuf,
                                           const float* __restrict__ b1,
                                           const float* __restrict__ b2,
                                           const float* __restrict__ b3,
                                           int lane, float (&dotp)[8], float (&n1p)[8],
                                           float (&n2p)[8]) {
  const int m = lane & 15, hi = lane >> 4, sel = hi ? 8 : 0;
  // constants for storing D-layout values straight into A-fragment layout:
  // value (row m'=r+hi*8, col o=nt*16+m) -> abuf[(nt>>1)*512 + readerLane*16 + j]
  // readerLane = m' + (m>=8 ? 16 : 0), j = (m&7) + 8*(nt&1)
  const int rl_off = hi * 8 + ((m & 8) ? 16 : 0);
  const int j_base = (m & 7);
  const v16bf* av = (const v16bf*)abuf + lane;   // this lane's chunk slots, stride 32 v16bf

  // ---------- layer 1: 128 -> 256, selu ----------
  v16bf A1[4];
  const float* rowp = S + (size_t)(row0 + m) * 128;
#pragma unroll
  for (int kc = 0; kc < 4; kc++) A1[kc] = a_frag_from_f32(rowp, kc * 32, sel);
#pragma unroll
  for (int nt = 0; nt < 16; nt++) {
    v8f c = mlp_tile<4>(A1, wlane + W1_OFS + nt * 4 * 512, b1[nt * 16 + m]);
#pragma unroll
    for (int r = 0; r < 8; r++)
      abuf[(nt >> 1) * 512 + (r + rl_off) * 16 + j_base + 8 * (nt & 1)] =
          (__bf16)selu_f(c[r]);
  }

  // ---------- layer 2: 256 -> 256, selu ----------
  v16bf A2[8];
#pragma unroll
  for (int kc = 0; kc < 8; kc++) A2[kc] = av[kc * 32];
#pragma unroll
  for (int nt = 0; nt < 16; nt++) {
    v8f c = mlp_tile<8>(A2, wlane + W2_OFS + nt * 8 * 512, b2[nt * 16 + m]);
#pragma unroll
    for (int r = 0; r < 8; r++)
      abuf[(nt >> 1) * 512 + (r + rl_off) * 16 + j_base + 8 * (nt & 1)] =
          (__bf16)selu_f(c[r]);
  }

  // ---------- layer 3: 256 -> 128, linear ----------
  v16bf A3[8];
#pragma unroll
  for (int kc = 0; kc < 8; kc++) A3[kc] = av[kc * 32];
#pragma unroll
  for (int nt = 0; nt < 8; nt++) {
    v8f c = mlp_tile<8>(A3, wlane + W3_OFS + nt * 8 * 512, b3[nt * 16 + m]);
    unsigned int* ep = ebuf + (nt * 16 + m) * 12 + hi * 4;   // col-major e1 stash, 48B/col
    if (!SECOND) {
      uint4 pk;
      pk.x = pack_bf16x2(c[0], c[1]);
      pk.y = pack_bf16x2(c[2], c[3]);
      pk.z = pack_bf16x2(c[4], c[5]);
      pk.w = pack_bf16x2(c[6], c[7]);
      *(uint4*)ep = pk;
    } else {
      uint4 pk = *(const uint4*)ep;
      unsigned int w[4] = {pk.x, pk.y, pk.z, pk.w};
#pragma unroll
      for (int r = 0; r < 8; r++) {
        float ev = bf16_to_f32((w[r >> 1] >> ((r & 1) * 16)) & 0xFFFFu);
        dotp[r] += ev * c[r];
        n1p[r]  += ev * ev;
        n2p[r]  += c[r] * c[r];
      }
    }
  }
}

// ---------------- prep: f32 weights -> bf16, swizzled into B-fragment blocks ----------------
// Block (nt,kc) = 1KB: lane L, j in [0,16): W[nt*16 + (L&15)][kc*32 + (L>>4)*8 + (j&7) + (j>>3)*16]
__global__ void pack_weights_kernel(const float* __restrict__ W1, const float* __restrict__ W2,
                                    const float* __restrict__ W3, unsigned short* __restrict__ dst) {
  int t = blockIdx.x * blockDim.x + threadIdx.x;
  if (t >= 131072) return;
  const float* W; int K, e, base;
  if (t < 32768)      { W = W1; K = 128; e = t;          base = W1_OFS; }
  else if (t < 98304) { W = W2; K = 256; e = t - 32768;  base = W2_OFS; }
  else                { W = W3; K = 256; e = t - 98304;  base = W3_OFS; }
  int blk = e >> 9, idx = e & 511, L = idx >> 4, j = idx & 15;
  int kch = K / 32;
  int nt = blk / kch, kc = blk % kch;
  int o = nt * 16 + (L & 15);
  int k = kc * 32 + (L >> 4) * 8 + (j & 7) + (j >> 3) * 16;
  dst[base + e] = bf16_rne_bits(W[o * K + k]);
}

// ---------------- main kernel ----------------
__global__ __launch_bounds__(WAVES * 32)
void embed_sim_kernel(const float* __restrict__ s1, const float* __restrict__ s2,
                      const float* __restrict__ b1, const float* __restrict__ b2,
                      const float* __restrict__ b3,
                      const unsigned short* __restrict__ wbf, float* __restrict__ out) {
  extern __shared__ char smem[];

  // ---- stage all bf16 weights (256KB) into LDS via CDNA5 async memory->LDS DMA ----
  {
#if defined(ASYNC_BUILTIN)
    v4i* gsrc = (v4i*)wbf;              // builtin prototype wants generic int4* args
    v4i* ldst = (v4i*)smem;
    for (int i = (int)threadIdx.x; i < WGT_BYTES / 16; i += (int)blockDim.x)
      __builtin_amdgcn_global_load_async_to_lds_b128(gsrc + i, ldst + i, 0, 0);
# if __has_builtin(__builtin_amdgcn_s_wait_asynccnt)
    __builtin_amdgcn_s_wait_asynccnt(0);
# else
    asm volatile("s_wait_asynccnt 0x0" ::: "memory");
# endif
#else
    unsigned lbase = (unsigned)(unsigned long long)(lds_char_t*)smem;
    unsigned long long gbase = (unsigned long long)wbf;
    for (int i = threadIdx.x * 16; i < WGT_BYTES; i += (int)blockDim.x * 16) {
      unsigned laddr = lbase + (unsigned)i;
      unsigned long long gaddr = gbase + (unsigned)i;
      asm volatile("global_load_async_to_lds_b128 %0, %1, off"
                   :: "v"(laddr), "v"(gaddr) : "memory");
    }
    asm volatile("s_wait_asynccnt 0x0" ::: "memory");
#endif
  }
  __syncthreads();

  const int wave = threadIdx.x >> 5, lane = threadIdx.x & 31;
  const int tile = blockIdx.x * WAVES + wave;
  const int row0 = tile * 16;
  if (row0 >= N_ROWS) return;

  const __bf16* wlane = (const __bf16*)smem + lane * 16;   // per-lane weight slot base
  __bf16*       abuf  = (__bf16*)(smem + WGT_BYTES + wave * WAVE_SCR);
  unsigned int* ebuf  = (unsigned int*)(smem + WGT_BYTES + wave * WAVE_SCR + ABUF_BYTES);

  float dotp[8], n1p[8], n2p[8];
#pragma unroll
  for (int r = 0; r < 8; r++) { dotp[r] = 0.f; n1p[r] = 0.f; n2p[r] = 0.f; }

  embed_pass<false>(s1, row0, wlane, abuf, ebuf, b1, b2, b3, lane, dotp, n1p, n2p);
  embed_pass<true >(s2, row0, wlane, abuf, ebuf, b1, b2, b3, lane, dotp, n1p, n2p);

  // reduce each row's partials across its 16-lane group (wave32 xor shuffles)
#pragma unroll
  for (int off = 1; off < 16; off <<= 1) {
#pragma unroll
    for (int r = 0; r < 8; r++) {
      dotp[r] += __shfl_xor(dotp[r], off, 32);
      n1p[r]  += __shfl_xor(n1p[r],  off, 32);
      n2p[r]  += __shfl_xor(n2p[r],  off, 32);
    }
  }
  const int m = lane & 15, hi = lane >> 4;
  if (m == 0) {   // lanes 0 (rows 0-7) and 16 (rows 8-15) write results
#pragma unroll
    for (int r = 0; r < 8; r++) {
      float d = dotp[r];
      float nn = sqrtf(n1p[r]) * sqrtf(n2p[r]);
      float o = d / nn;
      o = (o + 1.f) * 0.5f;
      float res = 1.f / (1.f + __expf(-o));
      __builtin_nontemporal_store(res, out + row0 + hi * 8 + r);
    }
  }
}

// ---------------- host entry ----------------
extern "C" void kernel_launch(void* const* d_in, const int* in_sizes, int n_in,
                              void* d_out, int out_size, void* d_ws, size_t ws_size,
                              hipStream_t stream) {
  const float* s1 = (const float*)d_in[0];
  const float* s2 = (const float*)d_in[1];
  const float* W1 = (const float*)d_in[2];
  const float* b1 = (const float*)d_in[3];
  const float* W2 = (const float*)d_in[4];
  const float* b2 = (const float*)d_in[5];
  const float* W3 = (const float*)d_in[6];
  const float* b3 = (const float*)d_in[7];
  float* out = (float*)d_out;
  unsigned short* wbf = (unsigned short*)d_ws;   // 256KB swizzled bf16 weights

  (void)hipFuncSetAttribute((const void*)embed_sim_kernel,
                            hipFuncAttributeMaxDynamicSharedMemorySize, SMEM_TOTAL);

  pack_weights_kernel<<<512, 256, 0, stream>>>(W1, W2, W3, wbf);

  const int blocks = N_ROWS / 16 / WAVES;   // 1024 blocks, 1 tile per wave
  embed_sim_kernel<<<blocks, WAVES * 32, SMEM_TOTAL, stream>>>(s1, s2, b1, b2, b3, wbf, out);
}
